// QuantizedBatchNorm3d_73684458930259
// MI455X (gfx1250) — compile-verified, matrix operand order checked by reading, not run on hardware
//
#include <hip/hip_runtime.h>
#include <hip/hip_bf16.h>

// QuantizedBatchNorm3d (training fwd) for x:(8,64,32,64,64) f32.
// fake_quant == f32 -> IEEE half (RNE) -> f32 with saturation at +-65504.
// Memory-bound: ~768MB traffic -> ~33us floor at 23.3 TB/s.
// Channel sums/sumsq computed with V_WMMA_F32_16X16X32_F16 against a ones-B
// matrix (x_q is exactly fp16, so products are exact, accum in f32).

typedef __attribute__((ext_vector_type(16))) _Float16 v16h;
typedef __attribute__((ext_vector_type(8)))  float    v8f;

#define N_    8
#define C_    64
#define DHW_  (32 * 64 * 64)          /* 131072 floats per (n,c) slab      */
#define TOTAL_ ((size_t)N_ * C_ * DHW_) /* 67108864 elements                */
#define PARTS_ 4                       /* reduce blocks per slab            */
#define CHUNK_ (DHW_ / PARTS_)         /* 32768 floats per reduce block     */
#define RITERS_ (CHUNK_ / 4096)        /* 8 iters; 8 waves * 512 floats/iter */

__device__ __forceinline__ float fq(float v) {
    // saturating f32 -> half(RNE) -> f32 ; matches reference fake_quant
    v = fminf(fmaxf(v, -65504.0f), 65504.0f);
    return (float)(_Float16)v;
}

__global__ void bn3dq_zero_ws(float* __restrict__ ws) {
    ws[threadIdx.x] = 0.0f;   // launched with 128 threads: sum[64], sumsq[64]
}

__global__ __launch_bounds__(256) void bn3dq_reduce(const float* __restrict__ x,
                                                    float* __restrict__ ws) {
    const int slab = blockIdx.x / PARTS_;          // n*C + c
    const int part = blockIdx.x % PARTS_;
    const int c    = slab & (C_ - 1);
    const int lane = threadIdx.x & 31;             // wave32
    const int wave = threadIdx.x >> 5;

    const float4* base =
        (const float4*)(x + (size_t)slab * DHW_ + (size_t)part * CHUNK_);

    v16h ones;
#pragma unroll
    for (int i = 0; i < 16; ++i) ones[i] = (_Float16)1.0f;

    v8f accS = {};   // row sums of x_q
    v8f accQ = {};   // row sums of x_q^2

    for (int it = 0; it < RITERS_; ++it) {
        // wave covers 128 contiguous float4 (512 floats); 4 coalesced b128 loads
        const float4* wp = base + it * 1024 + wave * 128;
        float4 f0 = wp[lane];
        float4 f1 = wp[lane + 32];
        float4 f2 = wp[lane + 64];
        float4 f3 = wp[lane + 96];
        float xs[16] = {f0.x, f0.y, f0.z, f0.w, f1.x, f1.y, f1.z, f1.w,
                        f2.x, f2.y, f2.z, f2.w, f3.x, f3.y, f3.z, f3.w};
        v16h a, a2;
#pragma unroll
        for (int i = 0; i < 16; ++i) {
            _Float16 h = (_Float16)fminf(fmaxf(xs[i], -65504.0f), 65504.0f);
            a[i]  = h;        // exact fp16 quantized value
            a2[i] = h * h;    // fp16 square: error << fake_quant(var) step
        }
        // D = A x Ones + C : every output column n holds rowsum(m) in f32.
        accS = __builtin_amdgcn_wmma_f32_16x16x32_f16(
            false, a,  false, ones, (short)0, accS, false, false);
        accQ = __builtin_amdgcn_wmma_f32_16x16x32_f16(
            false, a2, false, ones, (short)0, accQ, false, false);
    }

    // C/D layout: lanes 0-15 hold M=0..7 (VGPR0..7), lanes 16-31 hold M=8..15.
    float s = 0.0f, sq = 0.0f;
#pragma unroll
    for (int i = 0; i < 8; ++i) { s += accS[i]; sq += accQ[i]; }
    s  += __shfl_xor(s, 16);    // add the other half's rows (columns identical)
    sq += __shfl_xor(sq, 16);

    __shared__ float ls[8], lq[8];
    if (lane == 0) { ls[wave] = s; lq[wave] = sq; }
    __syncthreads();
    if (threadIdx.x == 0) {
        float S = 0.0f, Q = 0.0f;
#pragma unroll
        for (int w = 0; w < 8; ++w) { S += ls[w]; Q += lq[w]; }
        atomicAdd(&ws[c], S);          // global_atomic_add_f32
        atomicAdd(&ws[C_ + c], Q);
    }
}

__global__ void bn3dq_stats(const float* __restrict__ weight,
                            const float* __restrict__ bias,
                            float* __restrict__ ws) {
    const int c = threadIdx.x;                     // 64 threads
    const float invM = 1.0f / (float)((size_t)N_ * DHW_);
    float mean = ws[c] * invM;
    float var  = fmaxf(ws[C_ + c] * invM - mean * mean, 0.0f);
    float mq = fq(mean);
    float vq = fq(var);
    float wq = fq(weight[c]);
    float bq = fq(bias[c]);
    float inv = 1.0f / sqrtf(vq + 1e-5f);
    ws[2 * C_ + c] = mq;
    ws[3 * C_ + c] = inv;
    ws[4 * C_ + c] = wq;
    ws[5 * C_ + c] = bq;
}

__global__ __launch_bounds__(256) void bn3dq_apply(const float* __restrict__ x,
                                                   const float* __restrict__ ws,
                                                   float* __restrict__ out) {
    // 4096 floats per block, 32 blocks per (n,c) slab -> channel is uniform.
    const int c = (blockIdx.x >> 5) & (C_ - 1);
    const float mean = ws[2 * C_ + c];   // block-uniform -> scalar loads
    const float inv  = ws[3 * C_ + c];
    const float w    = ws[4 * C_ + c];
    const float b    = ws[5 * C_ + c];

    const size_t base = (size_t)blockIdx.x * 1024;  // in float4 units
    const float4* xp = (const float4*)x + base;
    float4* op = (float4*)out + base;
#pragma unroll
    for (int j = 0; j < 4; ++j) {
        const int idx = threadIdx.x + j * 256;
        float4 v = xp[idx];
        float4 r;
        r.x = fq(w * ((fq(v.x) - mean) * inv) + b);
        r.y = fq(w * ((fq(v.y) - mean) * inv) + b);
        r.z = fq(w * ((fq(v.z) - mean) * inv) + b);
        r.w = fq(w * ((fq(v.w) - mean) * inv) + b);
        op[idx] = r;
    }
}

extern "C" void kernel_launch(void* const* d_in, const int* in_sizes, int n_in,
                              void* d_out, int out_size, void* d_ws, size_t ws_size,
                              hipStream_t stream) {
    const float* x      = (const float*)d_in[0];
    const float* weight = (const float*)d_in[1];
    const float* bias   = (const float*)d_in[2];
    float* out = (float*)d_out;
    float* ws  = (float*)d_ws;   // needs 6*64 floats = 1.5 KB

    bn3dq_zero_ws<<<1, 128, 0, stream>>>(ws);
    bn3dq_reduce<<<N_ * C_ * PARTS_, 256, 0, stream>>>(x, ws);   // 2048 blocks
    bn3dq_stats<<<1, C_, 0, stream>>>(weight, bias, ws);
    bn3dq_apply<<<(int)(TOTAL_ / 4096), 256, 0, stream>>>(x, ws, out); // 16384 blocks
}
